// DIFFormerConv_82085414961441
// MI455X (gfx1250) — matrix operand order repaired and checked by reading, not compiled, not run on hardware
//
#include <hip/hip_runtime.h>
#include <hip/hip_bf16.h>

#define H 4
#define D 64
#define CIN 256
#define HD 256          // H*D
#define CHUNK 32        // nodes per tile-chunk
#define S1SPLIT 4       // sub-blocks per graph in stage1
#define S1CHUNKS 8      // chunks per stage1 block (S1SPLIT*S1CHUNKS*CHUNK >= Mmax=1000)

#define XPAD  (CIN + 2)    // row stride (ushort) for x / q bf16 tiles
#define TPAD  (CHUNK + 2)  // row stride (ushort) for transposed k/v tiles
#define FPAD  (2 * HD + 2) // row stride (float) for fp32 k|v tile

typedef __attribute__((ext_vector_type(16))) __bf16 v16bf;
typedef __attribute__((ext_vector_type(8))) float v8f;

union FragA { v16bf b; unsigned int d[8]; };

__device__ __forceinline__ unsigned short f2bf(float f) {
    union { float f; unsigned u; } x; x.f = f;
    unsigned r = x.u + 0x7FFFu + ((x.u >> 16) & 1u);   // round-to-nearest-even
    return (unsigned short)(r >> 16);
}
__device__ __forceinline__ unsigned int f2bf2(float lo, float hi) {
    return (unsigned int)f2bf(lo) | ((unsigned int)f2bf(hi) << 16);
}

// Load an A fragment (contiguous k order, same as the packed B) from a row-major ushort row:
// element e -> k = kbase + 16*half + e; pairs are dword-contiguous.
__device__ __forceinline__ void load_fragA(FragA& a, const unsigned short* row,
                                           int kbase, int half) {
    const unsigned short* p = row + kbase + 16 * half;
    #pragma unroll
    for (int j = 0; j < 8; ++j) {
        a.d[j]     = *(const unsigned int*)(p + 2 * j);
    }
}

// ---------------------------------------------------------------------------
// Kernel 0: pack Wq|Wk|Wv (fp32 [256,256] row-major, K x N) into bf16
// B-fragment layout [m][ct][ks][lane][16]: element e -> K = ks*32+16*half+e,
// col n = ct*16 + (lane&15).  One contiguous 32-byte load per lane.
// ---------------------------------------------------------------------------
__global__ void pack_w(const float* __restrict__ Wq, const float* __restrict__ Wk,
                       const float* __restrict__ Wv, unsigned short* __restrict__ wp) {
    int idx = blockIdx.x * 256 + threadIdx.x;
    if (idx >= 3 * 16 * 8 * 512) return;
    int e    = idx & 15;
    int lane = (idx >> 4) & 31;
    int ks   = (idx >> 9) & 7;
    int ct   = (idx >> 12) & 15;
    int m    = idx >> 16;
    int K = ks * 32 + 16 * (lane >> 4) + e;
    int n = ct * 16 + (lane & 15);
    const float* W = (m == 0) ? Wq : (m == 1) ? Wk : Wv;
    wp[idx] = f2bf(W[K * HD + n]);
}

// ---------------------------------------------------------------------------
// Stage 1: per-graph streaming of k,v; accumulate kv = k^T v (WMMA over node
// dim), k_sum (normalized k), v_sum.  Grid = B * S1SPLIT blocks of 256.
// ---------------------------------------------------------------------------
__global__ __launch_bounds__(256) void stage1(
    const float* __restrict__ x, const int* __restrict__ nn,
    const float* __restrict__ kb, const float* __restrict__ vb,
    const unsigned short* __restrict__ wp,
    float* __restrict__ kvg, float* __restrict__ ksum, float* __restrict__ vsum) {

    __shared__ unsigned short xbf[CHUNK][XPAD];    // x chunk, bf16 bits
    __shared__ float          kvf[CHUNK][FPAD];    // raw k|v fp32
    __shared__ unsigned short knT[HD][TPAD];       // normalized k, bf16, [col][node]
    __shared__ unsigned short vT[HD][TPAD];        // v, bf16, [col][node]
    __shared__ float          sums[2 * HD];        // k_sum | v_sum
    __shared__ float          rnorm[CHUNK][H];

    int b   = blockIdx.x / S1SPLIT;
    int s   = blockIdx.x % S1SPLIT;
    int tid = threadIdx.x;
    int lane = tid & 31, w = tid >> 5;
    int half = lane >> 4;

    long long start = 0;
    for (int i = 0; i < b; ++i) start += (long long)nn[i];
    int Mb = nn[b];

    sums[tid] = 0.0f; sums[tid + HD] = 0.0f;

    v8f kvacc[8];
    #pragma unroll
    for (int j = 0; j < 8; ++j)
        #pragma unroll
        for (int q = 0; q < 8; ++q) kvacc[j][q] = 0.0f;

    int ldr = tid >> 3;                 // this thread's load row
    int ldc = (tid & 7) * 32;           // and columns [ldc, ldc+32)

    for (int c = 0; c < S1CHUNKS; ++c) {
        int base  = (s * S1CHUNKS + c) * CHUNK;
        int valid = Mb - base; if (valid > CHUNK) valid = CHUNK;

        {   // load x chunk (float4) -> bf16 LDS, packed dword stores
            const float4* xr = (const float4*)(x + (start + base + ldr) * (long long)CIN + ldc);
            bool rv = (valid > 0) && (ldr < valid);
            #pragma unroll
            for (int j = 0; j < 8; ++j) {
                float4 f = rv ? xr[j] : float4{0.f, 0.f, 0.f, 0.f};
                *(unsigned int*)&xbf[ldr][ldc + 4 * j]     = f2bf2(f.x, f.y);
                *(unsigned int*)&xbf[ldr][ldc + 4 * j + 2] = f2bf2(f.z, f.w);
            }
        }
        // prefetch next chunk's x rows (global_prefetch_b8) to hide HBM latency
        if (c + 1 < S1CHUNKS) {
            int nbase = base + CHUNK;
            if (nbase + ldr < Mb)
                __builtin_prefetch(x + (start + nbase + ldr) * (long long)CIN + ldc, 0, 1);
        }
        __syncthreads();

        if (valid > 0) {     // block-uniform predicate
            // --- GEMM: [32 x 256] @ Wk|Wv -> kvf (64 tiles, 8 per wave) ---
            for (int i = 0; i < 8; ++i) {
                int t = w + 8 * i;
                int rt = t >> 5;          // 0..1
                int col = t & 31;         // 0..31 over k|v concat
                int m  = 1 + (col >> 4);  // 1=k, 2=v
                int ct = col & 15;
                v8f acc;
                #pragma unroll
                for (int q = 0; q < 8; ++q) acc[q] = 0.0f;
                const unsigned short* arow = &xbf[rt * 16 + (lane & 15)][0];
                const unsigned short* wpp = wp + (((m * 16 + ct) * 8) * 32 + lane) * 16;
                #pragma unroll
                for (int ks = 0; ks < 8; ++ks) {
                    FragA a, bf;
                    load_fragA(a, arow, ks * 32, half);
                    #pragma unroll
                    for (int j = 0; j < 8; ++j)
                        bf.d[j] = ((const unsigned int*)(wpp + ks * 512))[j];
                    acc = __builtin_amdgcn_wmma_f32_16x16x32_bf16(
                              false, a.b, false, bf.b, (short)0, acc, false, false);
                }
                int colg = ct * 16 + (lane & 15);
                float bias = (m == 1) ? kb[colg] : vb[colg];
                #pragma unroll
                for (int q = 0; q < 8; ++q) {
                    int row = rt * 16 + q + 8 * half;
                    kvf[row][(m - 1) * HD + colg] = acc[q] + bias;
                }
            }
            __syncthreads();

            // --- per-(row,head) L2 norm of k ---
            if (tid < CHUNK * H) {
                int r = tid >> 2, h = tid & 3;
                float sa = 0.0f;
                for (int d = 0; d < D; ++d) { float t2 = kvf[r][h * D + d]; sa += t2 * t2; }
                rnorm[r][h] = (r < valid && sa > 0.0f) ? rsqrtf(sa) : 0.0f;
            }
            __syncthreads();

            // --- normalize k, convert to bf16 transposed [col][node], sums ---
            {
                int ccol = tid; int h = ccol >> 6;
                float ka = 0.0f, va = 0.0f;
                #pragma unroll
                for (int r = 0; r < CHUNK; r += 2) {
                    float kn0 = kvf[r][ccol] * rnorm[r][h];
                    float kn1 = kvf[r + 1][ccol] * rnorm[r + 1][h];
                    float v0 = (r     < valid) ? kvf[r][HD + ccol]     : 0.0f;
                    float v1 = (r + 1 < valid) ? kvf[r + 1][HD + ccol] : 0.0f;
                    *(unsigned int*)&knT[ccol][r] = f2bf2(kn0, kn1);
                    *(unsigned int*)&vT[ccol][r]  = f2bf2(v0, v1);
                    ka += kn0 + kn1; va += v0 + v1;
                }
                sums[ccol] += ka; sums[HD + ccol] += va;
            }
            __syncthreads();

            // --- kv += k^T v : 64 tiles (h,dt,et), K = 32 nodes, 8 per wave ---
            for (int j = 0; j < 8; ++j) {
                int tt = w * 8 + j;
                int h = tt >> 4, dt = (tt >> 2) & 3, et = tt & 3;
                FragA a, bb;
                // A = k^T: row = d, K = node; transposed store -> contiguous
                load_fragA(a, &knT[h * D + dt * 16 + (lane & 15)][0], 0, half);
                // B = v: col = e, K = node = 16*half + e -> contiguous
                const unsigned short* vp = &vT[h * D + et * 16 + (lane & 15)][16 * half];
                #pragma unroll
                for (int q = 0; q < 8; ++q)
                    bb.d[q] = *(const unsigned int*)(vp + 2 * q);
                kvacc[j] = __builtin_amdgcn_wmma_f32_16x16x32_bf16(
                               false, a.b, false, bb.b, (short)0, kvacc[j], false, false);
            }
        }
        __syncthreads();
    }

    // flush partials with fp32 global atomics
    {
        for (int j = 0; j < 8; ++j) {
            int tt = w * 8 + j;
            int h = tt >> 4, dt = (tt >> 2) & 3, et = tt & 3;
            int ecol = et * 16 + (lane & 15);
            #pragma unroll
            for (int q = 0; q < 8; ++q) {
                int drow = dt * 16 + q + 8 * half;
                atomicAdd(&kvg[(((long long)b * H + h) * D + drow) * D + ecol], kvacc[j][q]);
            }
        }
        atomicAdd(&ksum[b * HD + tid], sums[tid]);
        atomicAdd(&vsum[b * HD + tid], sums[HD + tid]);
    }
}

// ---------------------------------------------------------------------------
// Pack kv (fp32 [B][H][64][64]) into bf16 B-fragments [b][h][ks2][et][lane][16]
// ---------------------------------------------------------------------------
__global__ void pack_kv(const float* __restrict__ kvg, unsigned short* __restrict__ kvp, int total) {
    int idx = blockIdx.x * 256 + threadIdx.x;
    if (idx >= total) return;
    int e    = idx & 15;
    int lane = (idx >> 4) & 31;
    int et   = (idx >> 9) & 3;
    int ks2  = (idx >> 11) & 1;
    int bh   = idx >> 12;           // b*H + h
    int Kd   = ks2 * 32 + 16 * (lane >> 4) + e;
    int ecol = et * 16 + (lane & 15);
    kvp[idx] = f2bf(kvg[((long long)bh * D + Kd) * D + ecol]);
}

// ---------------------------------------------------------------------------
// Stage 2: per 32-node chunk: q GEMM, normalize, numerator = qn @ kv + v_sum,
// den = qn . k_sum + Mb, out = mean over heads.  Grid = B*32 blocks of 256.
// ---------------------------------------------------------------------------
__global__ __launch_bounds__(256) void stage2(
    const float* __restrict__ x, const int* __restrict__ nn,
    const float* __restrict__ qb,
    const unsigned short* __restrict__ wp,
    const unsigned short* __restrict__ kvp,
    const float* __restrict__ ksum, const float* __restrict__ vsum,
    float* __restrict__ out) {

    __shared__ unsigned short xbf[CHUNK][XPAD];
    __shared__ float          qf[CHUNK][HD + 2];
    __shared__ unsigned short qbfs[CHUNK][XPAD];
    __shared__ float          rden[CHUNK][H];
    __shared__ float          rnq[CHUNK][H];
    __shared__ float          ksl[HD], vsl[HD];

    int b = blockIdx.x >> 5;
    int chunk = blockIdx.x & 31;
    int tid = threadIdx.x, lane = tid & 31, w = tid >> 5;
    int half = lane >> 4;

    long long start = 0;
    for (int i = 0; i < b; ++i) start += (long long)nn[i];
    int Mb = nn[b];
    int base = chunk * CHUNK;
    if (base >= Mb) return;                       // block-uniform
    int valid = Mb - base; if (valid > CHUNK) valid = CHUNK;

    ksl[tid] = ksum[b * HD + tid];
    vsl[tid] = vsum[b * HD + tid];
    {
        int r = tid >> 3; int c0 = (tid & 7) * 32;
        const float4* xr = (const float4*)(x + (start + base + r) * (long long)CIN + c0);
        bool rv = r < valid;
        #pragma unroll
        for (int j = 0; j < 8; ++j) {
            float4 f = rv ? xr[j] : float4{0.f, 0.f, 0.f, 0.f};
            *(unsigned int*)&xbf[r][c0 + 4 * j]     = f2bf2(f.x, f.y);
            *(unsigned int*)&xbf[r][c0 + 4 * j + 2] = f2bf2(f.z, f.w);
        }
    }
    __syncthreads();

    // --- q GEMM: 32 tiles, 4 per wave ---
    for (int i = 0; i < 4; ++i) {
        int t = w + 8 * i;
        int rt = t >> 4, ct = t & 15;
        v8f acc;
        #pragma unroll
        for (int q = 0; q < 8; ++q) acc[q] = 0.0f;
        const unsigned short* arow = &xbf[rt * 16 + (lane & 15)][0];
        const unsigned short* wpp = wp + ((ct * 8) * 32 + lane) * 16;   // m=0 (Wq)
        #pragma unroll
        for (int ks = 0; ks < 8; ++ks) {
            FragA a, bf;
            load_fragA(a, arow, ks * 32, half);
            #pragma unroll
            for (int j = 0; j < 8; ++j)
                bf.d[j] = ((const unsigned int*)(wpp + ks * 512))[j];
            acc = __builtin_amdgcn_wmma_f32_16x16x32_bf16(
                      false, a.b, false, bf.b, (short)0, acc, false, false);
        }
        int colg = ct * 16 + (lane & 15);
        float bias = qb[colg];
        #pragma unroll
        for (int q = 0; q < 8; ++q) {
            int row = rt * 16 + q + 8 * half;
            qf[row][colg] = acc[q] + bias;
        }
    }
    __syncthreads();

    if (tid < CHUNK * H) {
        int r = tid >> 2, h = tid & 3;
        float s2 = 0.0f, dt2 = 0.0f;
        for (int d = 0; d < D; ++d) {
            float qv = qf[r][h * D + d];
            s2 += qv * qv; dt2 += qv * ksl[h * D + d];
        }
        float rn = (s2 > 0.0f) ? rsqrtf(s2) : 0.0f;
        rnq[r][h] = rn;
        rden[r][h] = rn * dt2 + (float)Mb;
    }
    __syncthreads();
    {
        int ccol = tid; int h = ccol >> 6;
        #pragma unroll
        for (int r = 0; r < CHUNK; r += 2) {
            float q0 = qf[r][ccol] * rnq[r][h];
            float q1 = qf[r + 1][ccol] * rnq[r + 1][h];
            qbfs[r][ccol]     = f2bf(q0);
            qbfs[r + 1][ccol] = f2bf(q1);
        }
    }
    __syncthreads();

    // --- numerator + combine: wave w -> (rt = w>>2, et = w&3) ---
    {
        int rt = w >> 2, et = w & 3;
        int ecol = et * 16 + (lane & 15);
        float outacc[8];
        #pragma unroll
        for (int q = 0; q < 8; ++q) outacc[q] = 0.0f;
        const unsigned short* arow = &qbfs[rt * 16 + (lane & 15)][0];
        for (int h = 0; h < H; ++h) {
            v8f acc;
            #pragma unroll
            for (int q = 0; q < 8; ++q) acc[q] = 0.0f;
            #pragma unroll
            for (int ks2 = 0; ks2 < 2; ++ks2) {
                FragA a, bf;
                load_fragA(a, arow, h * D + ks2 * 32, half);
                const unsigned short* kp =
                    kvp + ((((long long)b * H + h) * 2 + ks2) * 4 + et) * 512 + lane * 16;
                #pragma unroll
                for (int j = 0; j < 8; ++j)
                    bf.d[j] = ((const unsigned int*)kp)[j];
                acc = __builtin_amdgcn_wmma_f32_16x16x32_bf16(
                          false, a.b, false, bf.b, (short)0, acc, false, false);
            }
            float vs = vsl[h * D + ecol];
            #pragma unroll
            for (int q = 0; q < 8; ++q) {
                int row = rt * 16 + q + 8 * half;
                outacc[q] += (acc[q] + vs) / rden[row][h];
            }
        }
        #pragma unroll
        for (int q = 0; q < 8; ++q) {
            int row = rt * 16 + q + 8 * half;
            if (row < valid) {
                long long node = start + base + row;
                out[node * D + ecol] = outacc[q] * 0.25f;
            }
        }
    }
}

// ---------------------------------------------------------------------------
extern "C" void kernel_launch(void* const* d_in, const int* in_sizes, int n_in,
                              void* d_out, int out_size, void* d_ws, size_t ws_size,
                              hipStream_t stream) {
    const float* x  = (const float*)d_in[0];
    const int*   nnp = (const int*)d_in[1];
    const float* Wq = (const float*)d_in[2];
    const float* qb = (const float*)d_in[3];
    const float* Wk = (const float*)d_in[4];
    const float* kb = (const float*)d_in[5];
    const float* Wv = (const float*)d_in[6];
    const float* vb = (const float*)d_in[7];
    float* out = (float*)d_out;
    int B = in_sizes[1];

    char* ws = (char*)d_ws;
    size_t wpack_bytes = 3 * 16 * 8 * 512 * sizeof(unsigned short);    // 393216
    size_t kv_bytes    = (size_t)B * H * D * D * sizeof(float);
    size_t sum_bytes   = (size_t)B * HD * sizeof(float);
    unsigned short* wpack = (unsigned short*)ws;
    float* kvg  = (float*)(ws + wpack_bytes);
    float* ksum = (float*)(ws + wpack_bytes + kv_bytes);
    float* vsum = (float*)(ws + wpack_bytes + kv_bytes + sum_bytes);
    unsigned short* kvp = (unsigned short*)(ws + wpack_bytes + kv_bytes + 2 * sum_bytes);

    // re-zero accumulators every call (graph-capture safe)
    hipMemsetAsync(kvg, 0, kv_bytes + 2 * sum_bytes, stream);

    pack_w<<<768, 256, 0, stream>>>(Wq, Wk, Wv, wpack);
    stage1<<<B * S1SPLIT, 256, 0, stream>>>(x, nnp, kb, vb, wpack, kvg, ksum, vsum);
    int kvtot = B * H * 2 * 4 * 512;
    pack_kv<<<(kvtot + 255) / 256, 256, 0, stream>>>(kvg, kvp, kvtot);
    stage2<<<B * 32, 256, 0, stream>>>(x, nnp, qb, wpack, kvp, ksum, vsum, out);
}
